// HCGNN_56985626083711
// MI455X (gfx1250) — compile-verified, matrix operand order checked by reading, not run on hardware
//
#include <hip/hip_runtime.h>
#include <hip/hip_bf16.h>

// ---------------------------------------------------------------------------
// HGT forward on gfx1250 (MI455X). All dense [N,64]@[64,64] projections run on
// v_wmma_f32_16x16x32_f16 (f16 in, f32 accum). Per-relation a_rel/m_rel
// einsums are folded into the K/V weights (block-diagonal 64x64) by a small
// fuse kernel, so every node-level op is a WMMA GEMM. Edge softmax/aggregation
// is the memory-bound scalar part (atomicMax-softmax + atomicAdd scatter).
//
// W is staged in LDS pre-swizzled into the exact WMMA B-fragment layout:
// 256 fragments (2 ksteps x 4 tiles x 32 lanes) of 16 halves; one thread
// builds one fragment, and reads become 2x ds_load_b128 per tile. All 4 B
// fragments of a kstep are hoisted before the WMMA chain so the loads clause
// together and the WMMAs drain with staged dscnt waits.
//
// Assumed d_in layout (setup_inputs() insertion order, params flattened as a
// jax pytree with sorted dict keys):
//   0 x_event[100000,32] 1 x_application[20000,16] 2 x_offer[30000,16]
//   3..8 ei_* int32 [2,E] in EDGE_TYPES order
//   9..  params: convs[0](45), convs[1](45), lin(6), lin_out(2)
//   conv leaf offsets: a:0-5  a_rel:6-11  k:12-17  m_rel:18-23  p_rel:24-29
//                      q:30-35  skip:36-38  v:39-44
//   (node-type dicts sorted: application, event, offer; {b,w} sorted b first;
//    *_rel sorted by ET_KEY string)
// ---------------------------------------------------------------------------

typedef __attribute__((ext_vector_type(16))) _Float16 v16h;
typedef __attribute__((ext_vector_type(8)))  float    v8f;

__device__ __forceinline__ float gelu_exact(float x) {
    return 0.5f * x * (1.0f + erff(x * 0.70710678118654752f));
}
// monotonic uint encoding of float for atomicMax
__device__ __forceinline__ unsigned enc_f(float f) {
    unsigned u = __float_as_uint(f);
    return (u & 0x80000000u) ? ~u : (u | 0x80000000u);
}
__device__ __forceinline__ float dec_f(unsigned k) {
    return (k & 0x80000000u) ? __uint_as_float(k & 0x7fffffffu)
                             : __uint_as_float(~k);
}

// ---------------------------------------------------------------------------
// Inner WMMA loop: A-frag per ISA 16-bit A 16x32 layout (two contiguous
// 8-half chunks per lane), B-frags preswizzled in LDS (one v16h per lane).
// GUARD=false is the wave-uniform interior fast path (no row clamps).
// ---------------------------------------------------------------------------
template <int KDIM, int AACT, bool GUARD>
__device__ __forceinline__ void gemm64_compute(const float* __restrict__ A,
                                               const v16h* __restrict__ ldsB,
                                               int r0, int M, int lane,
                                               v8f acc[4]) {
    const int mrow = lane & 15;
    const int koff = (lane >> 4) * 8;
    const int KS = (KDIM + 31) / 32;
#pragma unroll
    for (int ks = 0; ks < KS; ++ks) {
        const int k0 = ks * 32;
        const int m = r0 + mrow;
        bool mok = true;
        const float* Arow;
        if (GUARD) {
            mok = (m < M);
            Arow = A + (long)(mok ? m : 0) * KDIM;
        } else {
            Arow = A + (long)m * KDIM;
        }
        v16h af;
#pragma unroll
        for (int i = 0; i < 8; ++i) {
            const int k1 = k0 + koff + i;       // folds at compile time
            const int k2 = k0 + 16 + koff + i;  // for KDIM = 32 / 64
            float x1 = (k1 < KDIM) ? Arow[k1] : 0.0f;
            float x2 = (k2 < KDIM) ? Arow[k2] : 0.0f;
            if (GUARD && !mok) { x1 = 0.0f; x2 = 0.0f; }
            if (AACT == 1) { x1 = gelu_exact(x1); x2 = gelu_exact(x2); }
            af[i]     = (_Float16)x1;
            af[i + 8] = (_Float16)x2;
        }
        // hoist all 4 B fragments -> one clause of 8x ds_load_b128, then the
        // WMMA chain drains with staged dscnt waits
        const v16h bf0 = ldsB[(ks * 4 + 0) * 32 + lane];
        const v16h bf1 = ldsB[(ks * 4 + 1) * 32 + lane];
        const v16h bf2 = ldsB[(ks * 4 + 2) * 32 + lane];
        const v16h bf3 = ldsB[(ks * 4 + 3) * 32 + lane];
        acc[0] = __builtin_amdgcn_wmma_f32_16x16x32_f16(
            false, af, false, bf0, (short)0, acc[0], false, false);
        acc[1] = __builtin_amdgcn_wmma_f32_16x16x32_f16(
            false, af, false, bf1, (short)0, acc[1], false, false);
        acc[2] = __builtin_amdgcn_wmma_f32_16x16x32_f16(
            false, af, false, bf2, (short)0, acc[2], false, false);
        acc[3] = __builtin_amdgcn_wmma_f32_16x16x32_f16(
            false, af, false, bf3, (short)0, acc[3], false, false);
    }
}

// ---------------------------------------------------------------------------
// WMMA GEMM: C[M,64] = OOP( AACT(A)[M,KDIM] @ W[KDIM,64] + bias )
//   AACT: 0 none, 1 gelu       OOP: 0 none, 1 relu, 2 sigmoid-skip blend
// One wave -> 16 rows x 64 cols (4 WMMA tiles), 8 waves/block -> 128 rows.
// ---------------------------------------------------------------------------
template <int KDIM, int AACT, int OOP>
__global__ __launch_bounds__(256) void gemm64_kernel(
    const float* __restrict__ A, const float* __restrict__ W,
    const float* __restrict__ bias, float* __restrict__ C,
    const float* __restrict__ Hold, const float* __restrict__ skip, int M) {
    __shared__ v16h ldsB[256];  // [kstep][tile][lane] -> 16 halves, 8 KB
    const int tid = threadIdx.x;
    {   // one thread builds one pre-swizzled B fragment
        const int ks = tid >> 7, t = (tid >> 5) & 3, l = tid & 31;
        const int n = t * 16 + (l & 15);
        const int koff = (l >> 4) * 8;
        v16h frag;
#pragma unroll
        for (int i = 0; i < 16; ++i) {
            const int k = ks * 32 + ((i < 8) ? i : i + 8) + koff;
            frag[i] = (k < KDIM) ? (_Float16)W[k * 64 + n] : (_Float16)0.0f;
        }
        ldsB[tid] = frag;
    }
    __syncthreads();

    const int lane = tid & 31;
    const int wv   = tid >> 5;
    const int r0   = blockIdx.x * 128 + wv * 16;
    if (r0 >= M) return;  // wave-uniform exit; EXEC stays all-ones for WMMA

    const v8f zero = {0.f, 0.f, 0.f, 0.f, 0.f, 0.f, 0.f, 0.f};
    v8f acc[4] = {zero, zero, zero, zero};

    if (r0 + 16 <= M)  // wave-uniform branch
        gemm64_compute<KDIM, AACT, false>(A, ldsB, r0, M, lane, acc);
    else
        gemm64_compute<KDIM, AACT, true>(A, ldsB, r0, M, lane, acc);

    float sa = 0.0f;
    if (OOP == 2) sa = 1.0f / (1.0f + expf(-skip[0]));
    const int ncol = lane & 15;
#pragma unroll
    for (int t = 0; t < 4; ++t) {
        const int n = t * 16 + ncol;
#pragma unroll
        for (int r = 0; r < 8; ++r) {
            const int m = r0 + r + (lane >> 4) * 8;  // ISA 32-bit C/D layout
            if (m < M) {
                float v = acc[t][r] + bias[n];
                if (OOP == 1) v = fmaxf(v, 0.0f);
                if (OOP == 2)
                    v = sa * v + (1.0f - sa) * Hold[(long)m * 64 + n];
                C[(long)m * 64 + n] = v;
            }
        }
    }
}

// ---------------------------------------------------------------------------
// Fold a_rel/m_rel (HEADS x 16 x 16) into 64x64 weights + 64 bias:
//   Wf[i][h*16+e] = sum_d W[i][h*16+d] * R[h][d][e]; same for bias.
// 24 slots = 2 layers x 6 edge types x {k,v}. Slot stride 4160 floats.
// ---------------------------------------------------------------------------
struct FuseArgs { const float* W[24]; const float* B[24]; const float* R[24]; };

__global__ __launch_bounds__(256) void fuse_kernel(FuseArgs fa,
                                                   float* __restrict__ out) {
    const int slot = blockIdx.x;
    const float* W  = fa.W[slot];
    const float* Bb = fa.B[slot];
    const float* R  = fa.R[slot];
    float* Wo = out + (long)slot * 4160;
    float* Bo = Wo + 4096;
    for (int o = threadIdx.x; o < 4096; o += blockDim.x) {
        const int i = o >> 6, c = o & 63, h = c >> 4, e = c & 15;
        float acc = 0.f;
#pragma unroll
        for (int d = 0; d < 16; ++d)
            acc += W[i * 64 + h * 16 + d] * R[h * 256 + d * 16 + e];
        Wo[o] = acc;
    }
    for (int c = threadIdx.x; c < 64; c += blockDim.x) {
        const int h = c >> 4, e = c & 15;
        float acc = 0.f;
#pragma unroll
        for (int d = 0; d < 16; ++d)
            acc += Bb[h * 16 + d] * R[h * 256 + d * 16 + e];
        Bo[c] = acc;
    }
}

// ---------------------------------------------------------------------------
// Edge phase kernels (per edge type). ei: [2,E], row0 = src, row1 = dst.
// ---------------------------------------------------------------------------
__global__ __launch_bounds__(256) void edge_alpha_kernel(
    const int* __restrict__ ei, int E, const float* __restrict__ Q,
    const float* __restrict__ Kp, const float* __restrict__ prel,
    float* __restrict__ alpha, unsigned* __restrict__ amax) {
    const int tid = blockIdx.x * blockDim.x + threadIdx.x;
    if (tid >= E * 4) return;
    const int e = tid >> 2, h = tid & 3;
    const int s = ei[e], d = ei[E + e];
    const float* q = Q  + (long)d * 64 + h * 16;
    const float* k = Kp + (long)s * 64 + h * 16;
    float acc = 0.f;
#pragma unroll
    for (int i = 0; i < 16; ++i) acc += q[i] * k[i];
    acc *= prel[h] * 0.25f;  // / sqrt(DH=16)
    alpha[tid] = acc;
    atomicMax(&amax[(long)d * 4 + h], enc_f(acc));
}

__global__ __launch_bounds__(256) void edge_exp_kernel(
    const int* __restrict__ ei, int E, const unsigned* __restrict__ amax,
    float* __restrict__ alpha, float* __restrict__ den) {
    const int tid = blockIdx.x * blockDim.x + threadIdx.x;
    if (tid >= E * 4) return;
    const int e = tid >> 2, h = tid & 3;
    const int d = ei[E + e];
    float am = dec_f(amax[(long)d * 4 + h]);
    if (!__builtin_isfinite(am)) am = 0.f;
    const float ex = expf(alpha[tid] - am);
    alpha[tid] = ex;
    atomicAdd(&den[(long)d * 4 + h], ex);
}

__global__ __launch_bounds__(256) void edge_scatter_kernel(
    const int* __restrict__ ei, int E, const float* __restrict__ Vp,
    const float* __restrict__ alpha, const float* __restrict__ den,
    float* __restrict__ out) {
    const long tid = (long)blockIdx.x * blockDim.x + threadIdx.x;
    if (tid >= (long)E * 64) return;
    const int e = (int)(tid >> 6), c = (int)(tid & 63), h = c >> 4;
    const int s = ei[e], d = ei[E + e];
    const float w = alpha[e * 4 + h] / (den[(long)d * 4 + h] + 1e-16f);
    atomicAdd(&out[(long)d * 64 + c], Vp[(long)s * 64 + c] * w);
}

__global__ __launch_bounds__(256) void fill_u32_kernel(unsigned* __restrict__ p,
                                                       unsigned v, long n) {
    const long i = (long)blockIdx.x * blockDim.x + threadIdx.x;
    if (i < n) p[i] = v;
}
__global__ __launch_bounds__(256) void fill_f32_kernel(float* __restrict__ p,
                                                       float v, long n) {
    const long i = (long)blockIdx.x * blockDim.x + threadIdx.x;
    if (i < n) p[i] = v;
}

__global__ __launch_bounds__(256) void final_out_kernel(
    const float* __restrict__ h, const float* __restrict__ w,
    const float* __restrict__ b, float* __restrict__ out, int M) {
    const int m = blockIdx.x * blockDim.x + threadIdx.x;
    if (m >= M) return;
    float acc = b[0];
    const float* row = h + (long)m * 64;
#pragma unroll
    for (int i = 0; i < 64; ++i) acc += row[i] * w[i];
    out[m] = acc;
}

// ---------------------------------------------------------------------------
extern "C" void kernel_launch(void* const* d_in, const int* in_sizes, int n_in,
                              void* d_out, int out_size, void* d_ws,
                              size_t ws_size, hipStream_t stream) {
    (void)n_in; (void)out_size; (void)ws_size;
    auto F = [&](int i) { return (const float*)d_in[i]; };
    auto I = [&](int i) { return (const int*)d_in[i]; };

    // node types: 0=event 1=application 2=offer
    const int IN_[3]  = {32, 16, 16};
    int Nnt[3];
    for (int nt = 0; nt < 3; ++nt) Nnt[nt] = in_sizes[nt] / IN_[nt];
    const int pidx[3] = {1, 0, 2};  // nt -> sorted param position (app,evt,off)

    // edge types (EDGE_TYPES order)
    const int etei[6]   = {3, 4, 5, 6, 7, 8};
    const int etsrel[6] = {2, 3, 4, 0, 1, 5};  // sorted a_rel/m_rel/p_rel slot
    const int etsrc[6]  = {0, 0, 0, 1, 1, 2};
    const int etdst[6]  = {0, 1, 2, 1, 0, 0};
    int etE[6];
    for (int e = 0; e < 6; ++e) etE[e] = in_sizes[etei[e]] / 2;

    const int convb[2] = {9, 54};
    const int linb = 99, loutb = 105;

    // ---- workspace carve-out ----
    size_t off = 0;
    auto alloc = [&](size_t bytes) -> void* {
        off = (off + 255) & ~(size_t)255;
        void* p = (char*)d_ws + off;
        off += bytes;
        return p;
    };
    float *hA[3], *hB[3], *Qb[3], *Ob[3];
    for (int nt = 0; nt < 3; ++nt) hA[nt] = (float*)alloc((size_t)Nnt[nt] * 64 * 4);
    for (int nt = 0; nt < 3; ++nt) hB[nt] = (float*)alloc((size_t)Nnt[nt] * 64 * 4);
    for (int nt = 0; nt < 3; ++nt) Qb[nt] = (float*)alloc((size_t)Nnt[nt] * 64 * 4);
    for (int nt = 0; nt < 3; ++nt) Ob[nt] = (float*)alloc((size_t)Nnt[nt] * 64 * 4);
    int maxN = Nnt[0], maxE = etE[0];
    for (int i = 1; i < 3; ++i) maxN = Nnt[i] > maxN ? Nnt[i] : maxN;
    for (int i = 1; i < 6; ++i) maxE = etE[i] > maxE ? etE[i] : maxE;
    float*    Kp    = (float*)alloc((size_t)maxN * 64 * 4);
    float*    Vp    = (float*)alloc((size_t)maxN * 64 * 4);
    float*    aBuf  = (float*)alloc((size_t)maxE * 4 * 4);
    unsigned* amax  = (unsigned*)alloc((size_t)maxN * 4 * 4);
    float*    den   = (float*)alloc((size_t)maxN * 4 * 4);
    float*    fused = (float*)alloc((size_t)24 * 4160 * 4);

    auto cdiv = [](long a, long b) { return (unsigned)((a + b - 1) / b); };

    // ---- 1. fold a_rel/m_rel into K/V weights ----
    FuseArgs fa;
    {
        int slot = 0;
        for (int l = 0; l < 2; ++l)
            for (int et = 0; et < 6; ++et)
                for (int kv = 0; kv < 2; ++kv) {
                    const int p    = pidx[etsrc[et]];
                    const int base = convb[l];
                    const int woff = (kv ? 39 : 12) + p * 2;  // 'v' : 'k'
                    fa.B[slot] = F(base + woff);
                    fa.W[slot] = F(base + woff + 1);
                    fa.R[slot] = F(base + (kv ? 18 : 6) + etsrel[et]);
                    ++slot;
                }
    }
    fuse_kernel<<<24, 256, 0, stream>>>(fa, fused);

    // ---- 2. input linear + relu ----
    gemm64_kernel<32, 0, 1><<<cdiv(Nnt[0], 128), 256, 0, stream>>>(
        F(0), F(linb + 3), F(linb + 2), hA[0], nullptr, nullptr, Nnt[0]);  // event
    gemm64_kernel<16, 0, 1><<<cdiv(Nnt[1], 128), 256, 0, stream>>>(
        F(1), F(linb + 1), F(linb + 0), hA[1], nullptr, nullptr, Nnt[1]);  // app
    gemm64_kernel<16, 0, 1><<<cdiv(Nnt[2], 128), 256, 0, stream>>>(
        F(2), F(linb + 5), F(linb + 4), hA[2], nullptr, nullptr, Nnt[2]);  // offer

    // ---- 3. HGT conv layers ----
    float* hc[3] = {hA[0], hA[1], hA[2]};
    float* hn[3] = {hB[0], hB[1], hB[2]};
    for (int l = 0; l < 2; ++l) {
        const int base = convb[l];
        // Q projections + zero message accumulators
        for (int nt = 0; nt < 3; ++nt) {
            const int p = pidx[nt];
            gemm64_kernel<64, 0, 0><<<cdiv(Nnt[nt], 128), 256, 0, stream>>>(
                hc[nt], F(base + 30 + p * 2 + 1), F(base + 30 + p * 2), Qb[nt],
                nullptr, nullptr, Nnt[nt]);
            fill_f32_kernel<<<cdiv((long)Nnt[nt] * 64, 256), 256, 0, stream>>>(
                Ob[nt], 0.0f, (long)Nnt[nt] * 64);
        }
        // per edge type: K' / V' via fused weights, then 3-pass softmax+scatter
        for (int et = 0; et < 6; ++et) {
            const int s = etsrc[et], d = etdst[et], E = etE[et];
            const int slotK = (l * 6 + et) * 2 + 0;
            const int slotV = (l * 6 + et) * 2 + 1;
            const float* WfK = fused + (long)slotK * 4160;
            const float* WfV = fused + (long)slotV * 4160;
            gemm64_kernel<64, 0, 0><<<cdiv(Nnt[s], 128), 256, 0, stream>>>(
                hc[s], WfK, WfK + 4096, Kp, nullptr, nullptr, Nnt[s]);
            gemm64_kernel<64, 0, 0><<<cdiv(Nnt[s], 128), 256, 0, stream>>>(
                hc[s], WfV, WfV + 4096, Vp, nullptr, nullptr, Nnt[s]);
            fill_u32_kernel<<<cdiv((long)Nnt[d] * 4, 256), 256, 0, stream>>>(
                amax, 0x007FFFFFu /* enc(-inf) */, (long)Nnt[d] * 4);
            fill_f32_kernel<<<cdiv((long)Nnt[d] * 4, 256), 256, 0, stream>>>(
                den, 0.0f, (long)Nnt[d] * 4);
            const int* ei = I(etei[et]);
            const float* prel = F(base + 24 + etsrel[et]);
            edge_alpha_kernel<<<cdiv((long)E * 4, 256), 256, 0, stream>>>(
                ei, E, Qb[d], Kp, prel, aBuf, amax);
            edge_exp_kernel<<<cdiv((long)E * 4, 256), 256, 0, stream>>>(
                ei, E, amax, aBuf, den);
            edge_scatter_kernel<<<cdiv((long)E * 64, 256), 256, 0, stream>>>(
                ei, E, Vp, aBuf, den, Ob[d]);
        }
        // node update: h_new = sig(skip)*(a_proj(gelu(out))) + (1-sig)*h
        for (int nt = 0; nt < 3; ++nt) {
            const int p = pidx[nt];
            gemm64_kernel<64, 1, 2><<<cdiv(Nnt[nt], 128), 256, 0, stream>>>(
                Ob[nt], F(base + p * 2 + 1), F(base + p * 2), hn[nt], hc[nt],
                F(base + 36 + p), Nnt[nt]);
        }
        for (int nt = 0; nt < 3; ++nt) {
            float* t = hc[nt]; hc[nt] = hn[nt]; hn[nt] = t;
        }
    }

    // ---- 4. output head: event @ [64,1] + b ----
    final_out_kernel<<<cdiv(Nnt[0], 256), 256, 0, stream>>>(
        hc[0], F(loutb + 1), F(loutb), (float*)d_out, Nnt[0]);
}
